// AttentionLayer_10591389352529
// MI455X (gfx1250) — matrix operand order repaired and checked by reading, not compiled, not run on hardware
//
#include <hip/hip_runtime.h>
#include <hip/hip_bf16.h>

// ---------------------------------------------------------------------------
// Local-window attention (FBLA), restructured:
//   qk  = q_feat @ (q_w^T k_w) + q_b @ k_w            (WMMA bf16 GEMM)
//   s_k = qk . local_k            (q.k_b is constant over k -> softmax-inv)
//   a   = softmax_k(s_k);  y = sum_k a_k * local_k    (gather kernel, fp32)
//   out = q_feat + y @ v_w^T + v_b                    (WMMA bf16 GEMM)
// Shapes: B=2, N=8192, C=128, window 3x3x3 (K=27), vol 48^3, edge pad=clamp.
// ---------------------------------------------------------------------------

typedef __attribute__((ext_vector_type(16))) __bf16 v16bf;
typedef __attribute__((ext_vector_type(8)))  float  v8f;

static constexpr int C  = 128;
static constexpr int NQ = 8192;          // queries per batch
static constexpr int S  = 48;
static constexpr int S2 = S * S;
static constexpr int S3 = S * S * S;

__device__ __forceinline__ unsigned short f2bf_bits(float f) {
  unsigned u = __builtin_bit_cast(unsigned, f);
  u = (u + 0x7FFFu + ((u >> 16) & 1u)) >> 16;   // round-to-nearest-even
  return (unsigned short)u;
}
__device__ __forceinline__ __bf16 bfbits(unsigned short s) {
  return __builtin_bit_cast(__bf16, s);
}
__device__ __forceinline__ int iclamp(int v, int lo, int hi) {
  return v < lo ? lo : (v > hi ? hi : v);
}

// ---------------------------------------------------------------------------
// Prep: WqkT[c'][c] = sum_d q_w[d][c] * k_w[d][c']  (stored bf16, row = c')
//       bqk[c']    = sum_d q_b[d] * k_w[d][c']
//       vw_bf      = bf16(v_w)  (row-major [d][c], already the B layout)
// ---------------------------------------------------------------------------
__global__ void prep_weights(const float* __restrict__ qw,
                             const float* __restrict__ qb,
                             const float* __restrict__ kw,
                             const float* __restrict__ vw,
                             unsigned short* __restrict__ wqkT_bf,
                             unsigned short* __restrict__ vw_bf,
                             float* __restrict__ bqk) {
  const int c  = threadIdx.x;   // input channel
  const int cp = blockIdx.x;    // output channel (c')
  float acc = 0.f;
  for (int d = 0; d < C; ++d) acc += qw[d * C + c] * kw[d * C + cp];
  wqkT_bf[cp * C + c] = f2bf_bits(acc);
  vw_bf[cp * C + c]   = f2bf_bits(vw[cp * C + c]);
  if (c == 0) {
    float s = 0.f;
    for (int d = 0; d < C; ++d) s += qb[d] * kw[d * C + cp];
    bqk[cp] = s;
  }
}

__global__ void cvt_f32_to_bf16(const float* __restrict__ src,
                                unsigned short* __restrict__ dst, int n) {
  int i = blockIdx.x * blockDim.x + threadIdx.x;
  if (i < n) dst[i] = f2bf_bits(src[i]);
}

// ---------------------------------------------------------------------------
// WMMA fragment loaders (wave32; A/B share contiguous k = 16h + e, not the ISA table order)
// A 16x32 bf16: lane m = L&15, kbase = (L>>4)*16; elem e -> K=kbase+e.
// B 32x16 bf16: lane n = L&15, kbase = (L>>4)*16; elem e -> K=kbase+e.
// C/D 16x16 f32: VGPR r -> M = r + (L>>4)*8, N = L&15.
// ---------------------------------------------------------------------------
__device__ __forceinline__ v16bf load_a_frag(const unsigned short* row0,
                                             int lane, int f) {
  const int m  = lane & 15;
  const int kb = (lane >> 4) * 16;
  const unsigned short* p = row0 + m * C + f * 32 + kb;
  v16bf a;
#pragma unroll
  for (int e = 0; e < 8; ++e) a[e] = bfbits(p[e]);
#pragma unroll
  for (int e = 0; e < 8; ++e) a[8 + e] = bfbits(p[8 + e]);
  return a;
}

__device__ __forceinline__ v16bf load_b_frag(const unsigned short* w_nk,
                                             int lane, int f, int j) {
  const int n  = j * 16 + (lane & 15);
  const int kb = (lane >> 4) * 16;
  const unsigned short* p = w_nk + n * C + f * 32 + kb;
  v16bf b;
#pragma unroll
  for (int e = 0; e < 16; ++e) b[e] = bfbits(p[e]);
  return b;
}

// ---------------------------------------------------------------------------
// GEMM 1: qk[rows][C] = qf_bf[rows][C] @ WqkT^T ... i.e. out[m][n] =
//         sum_k qf[m][k] * WqkT[n][k], + bqk[n].   One wave per 16-row tile.
// ---------------------------------------------------------------------------
__global__ __launch_bounds__(256) void qk_gemm(
    const unsigned short* __restrict__ qf_bf,
    const unsigned short* __restrict__ wqkT_bf,
    const float* __restrict__ bqk,
    float* __restrict__ qk, int rows) {
  const int lane = threadIdx.x & 31;
  const int wave = threadIdx.x >> 5;
  const int row0 = (blockIdx.x * (blockDim.x >> 5) + wave) * 16;
  if (row0 >= rows) return;

  v16bf A[4];
#pragma unroll
  for (int f = 0; f < 4; ++f)
    A[f] = load_a_frag(qf_bf + (size_t)row0 * C, lane, f);

#pragma unroll
  for (int j = 0; j < 8; ++j) {
    v8f acc = {};
#pragma unroll
    for (int f = 0; f < 4; ++f) {
      v16bf Bf = load_b_frag(wqkT_bf, lane, f, j);
      acc = __builtin_amdgcn_wmma_f32_16x16x32_bf16(
          false, A[f], false, Bf, (short)0, acc, false, false);
    }
    const int n = j * 16 + (lane & 15);
    const float bias = bqk[n];
#pragma unroll
    for (int r = 0; r < 8; ++r) {
      const int m = (lane >> 4) * 8 + r;
      qk[(size_t)(row0 + m) * C + n] = acc[r] + bias;
    }
  }
}

// ---------------------------------------------------------------------------
// Gather + softmax + window reduce. One wave per query; lane owns 4 channels.
// feat is channel-major (b,c,z,y,x); volume (113 MB) lives in the 192 MB L2.
// ---------------------------------------------------------------------------
__global__ __launch_bounds__(256) void gather_attn(
    const float* __restrict__ feat,
    const int*   __restrict__ proj,
    const float* __restrict__ qk,
    unsigned short* __restrict__ y_bf, int rows) {
  const int lane = threadIdx.x & 31;
  const int wave = threadIdx.x >> 5;
  const int g = blockIdx.x * (blockDim.x >> 5) + wave;
  if (g >= rows) return;

  const int b  = g / NQ;
  const int cz = proj[(size_t)g * 3 + 0];
  const int cy = proj[(size_t)g * 3 + 1];
  const int cx = proj[(size_t)g * 3 + 2];

  const float4 qv = *(const float4*)(qk + (size_t)g * C + lane * 4);
  const float* fb = feat + ((size_t)b * C + (size_t)lane * 4) * S3;

  float fv[27][4];
  float sc[27];
#pragma unroll
  for (int k = 0; k < 27; ++k) {
    const int z = iclamp(cz + (k / 9) - 1, 0, S - 1);
    const int y = iclamp(cy + (k / 3) % 3 - 1, 0, S - 1);
    const int x = iclamp(cx + (k % 3) - 1, 0, S - 1);
    const float* fp = fb + (size_t)z * S2 + y * S + x;
    const float f0 = fp[0];
    const float f1 = fp[(size_t)S3];
    const float f2 = fp[(size_t)2 * S3];
    const float f3 = fp[(size_t)3 * S3];
    fv[k][0] = f0; fv[k][1] = f1; fv[k][2] = f2; fv[k][3] = f3;
    sc[k] = qv.x * f0 + qv.y * f1 + qv.z * f2 + qv.w * f3;
  }

  // all-reduce each tap score across the wave (32 lanes x 4 ch = 128 ch)
#pragma unroll
  for (int k = 0; k < 27; ++k) {
    float v = sc[k];
#pragma unroll
    for (int off = 16; off >= 1; off >>= 1) v += __shfl_xor(v, off, 32);
    sc[k] = v;
  }

  float mx = sc[0];
#pragma unroll
  for (int k = 1; k < 27; ++k) mx = fmaxf(mx, sc[k]);
  float sum = 0.f;
#pragma unroll
  for (int k = 0; k < 27; ++k) { sc[k] = __expf(sc[k] - mx); sum += sc[k]; }
  const float inv = 1.0f / sum;

  float4 acc = {0.f, 0.f, 0.f, 0.f};
#pragma unroll
  for (int k = 0; k < 27; ++k) {
    const float w = sc[k] * inv;
    acc.x += w * fv[k][0]; acc.y += w * fv[k][1];
    acc.z += w * fv[k][2]; acc.w += w * fv[k][3];
  }

  unsigned short* yo = y_bf + (size_t)g * C + lane * 4;
  yo[0] = f2bf_bits(acc.x); yo[1] = f2bf_bits(acc.y);
  yo[2] = f2bf_bits(acc.z); yo[3] = f2bf_bits(acc.w);
}

// ---------------------------------------------------------------------------
// GEMM 2: out = q_feat + y @ v_w^T + v_b   (v_w row-major is B's [n][k])
// ---------------------------------------------------------------------------
__global__ __launch_bounds__(256) void out_gemm(
    const unsigned short* __restrict__ y_bf,
    const unsigned short* __restrict__ vw_bf,
    const float* __restrict__ vb,
    const float* __restrict__ q_feat,
    float* __restrict__ out, int rows) {
  const int lane = threadIdx.x & 31;
  const int wave = threadIdx.x >> 5;
  const int row0 = (blockIdx.x * (blockDim.x >> 5) + wave) * 16;
  if (row0 >= rows) return;

  v16bf A[4];
#pragma unroll
  for (int f = 0; f < 4; ++f)
    A[f] = load_a_frag(y_bf + (size_t)row0 * C, lane, f);

#pragma unroll
  for (int j = 0; j < 8; ++j) {
    v8f acc = {};
#pragma unroll
    for (int f = 0; f < 4; ++f) {
      v16bf Bf = load_b_frag(vw_bf, lane, f, j);
      acc = __builtin_amdgcn_wmma_f32_16x16x32_bf16(
          false, A[f], false, Bf, (short)0, acc, false, false);
    }
    const int n = j * 16 + (lane & 15);
    const float bias = vb[n];
#pragma unroll
    for (int r = 0; r < 8; ++r) {
      const int m = (lane >> 4) * 8 + r;
      const size_t idx = (size_t)(row0 + m) * C + n;
      out[idx] = acc[r] + bias + q_feat[idx];
    }
  }
}

// ---------------------------------------------------------------------------
extern "C" void kernel_launch(void* const* d_in, const int* in_sizes, int n_in,
                              void* d_out, int out_size, void* d_ws, size_t ws_size,
                              hipStream_t stream) {
  const float* q_feat = (const float*)d_in[0];
  const float* feat   = (const float*)d_in[1];
  const int*   proj   = (const int*)d_in[2];
  // d_in[3] hr_coord: unused by the reference math.
  const float* q_w = (const float*)d_in[4];
  const float* q_b = (const float*)d_in[5];
  const float* k_w = (const float*)d_in[6];
  // d_in[7] k_b: softmax-invariant (constant over window), unused.
  const float* v_w = (const float*)d_in[8];
  const float* v_b = (const float*)d_in[9];
  float* out = (float*)d_out;

  const int BN = in_sizes[0] / C;   // B*N = 16384

  // workspace carve-out
  char* w = (char*)d_ws;
  size_t o = 0;
  float*          qk_ws   = (float*)(w + o);          o += (size_t)BN * C * 4;
  unsigned short* y_bf    = (unsigned short*)(w + o); o += (size_t)BN * C * 2;
  unsigned short* qf_bf   = (unsigned short*)(w + o); o += (size_t)BN * C * 2;
  unsigned short* wqkT_bf = (unsigned short*)(w + o); o += (size_t)C * C * 2;
  unsigned short* vw_bf   = (unsigned short*)(w + o); o += (size_t)C * C * 2;
  float*          bqk     = (float*)(w + o);          o += (size_t)C * 4;
  (void)ws_size; (void)n_in; (void)out_size;

  // 1. fold weights: Wqk = q_w^T k_w (transposed, bf16), bqk, bf16(v_w)
  prep_weights<<<C, C, 0, stream>>>(q_w, q_b, k_w, v_w, wqkT_bf, vw_bf, bqk);

  // 2. q_feat -> bf16 for WMMA A-operands
  const int nq = BN * C;
  cvt_f32_to_bf16<<<(nq + 255) / 256, 256, 0, stream>>>(q_feat, qf_bf, nq);

  // 3. qk = q_feat @ Wqk + bqk   (bf16 WMMA, fp32 accum)
  const int tiles  = (BN + 15) / 16;
  const int blocks = (tiles + 7) / 8;    // 8 waves (tiles) per 256-thread block
  qk_gemm<<<blocks, 256, 0, stream>>>(qf_bf, wqkT_bf, bqk, qk_ws, BN);

  // 4. gather 27 taps, softmax, window-reduce -> y (bf16)
  gather_attn<<<(BN + 7) / 8, 256, 0, stream>>>(feat, proj, qk_ws, y_bf, BN);

  // 5. out = q_feat + y @ v_w^T + v_b   (bf16 WMMA, fp32 accum)
  out_gemm<<<blocks, 256, 0, stream>>>(y_bf, vw_bf, v_b, q_feat, out, BN);
}
